// attention_equ_42090679501098
// MI455X (gfx1250) — compile-verified, matrix operand order checked by reading, not run on hardware
//
#include <hip/hip_runtime.h>

// CDNA5 / gfx1250: wave32, WMMA 16x16x32 bf16 with f32 accumulate.
typedef __attribute__((ext_vector_type(16))) __bf16 v16bf;
typedef __attribute__((ext_vector_type(8)))  float  v8f;

union FragBF { v16bf v; unsigned int u[8]; };

// f32 -> bf16 (round-half-up), low 16 bits of return
__device__ __forceinline__ unsigned int bf16_1(float f) {
    return (__float_as_uint(f) + 0x8000u) >> 16;
}
// pack two f32 -> two bf16 in one dword (lo = even-k element), 3 VALU via v_perm_b32
__device__ __forceinline__ unsigned int bf16x2(float lo, float hi) {
    unsigned int ul = __float_as_uint(lo) + 0x8000u;
    unsigned int uh = __float_as_uint(hi) + 0x8000u;
    // result bytes [3,2,1,0] = {uh.b3, uh.b2, ul.b3, ul.b2}; a=uh holds bytes 4-7
    return __builtin_amdgcn_perm(uh, ul, 0x07060302u);
}

// Build a 16x32 bf16 A-fragment (or 32x16 B-fragment, symmetric assumption)
// from an LDS row of 16 packed k-pair dwords, per CDNA5 ISA 7.12.2:
//   lane half 0: K 0..7 in VGPR0-3, K 16..23 in VGPR4-7
//   lane half 1: K 8..15 in VGPR0-3, K 24..31 in VGPR4-7
// Each half is 4+4 contiguous dwords -> two ds_load_b128.
__device__ __forceinline__ v16bf gather_frag(const unsigned int* rowp, int half) {
    FragBF f;
#pragma unroll
    for (int p = 0; p < 8; ++p)
        f.u[p] = rowp[(p & 3) + (half << 2) + ((p >> 2) << 3)];
    return f.v;
}

// ---------------------------------------------------------------------------
// Kernel A: V = bf16(equ @ Wv^T), written per (e,b,h) as a (m=1024) x 96 matrix
// v_ws[(((e*8+b)*8+h)*1024 + m)*96 + l*32 + c]
// Tile: 128 rows x 64 cols, K=256 in chunks of 32. 8 waves, 4 subtiles/wave.
// ---------------------------------------------------------------------------
__global__ __launch_bounds__(256) void kA_v_gemm(
    const float* __restrict__ equ1, const float* __restrict__ equ2,
    const float* __restrict__ Wv, unsigned short* __restrict__ v_ws)
{
    __shared__ unsigned int a_lds[128 * 16];  // 128 rows x 16 k-pairs (bf16x2)
    __shared__ unsigned int b_lds[64 * 16];   // 64 cols  x 16 k-pairs

    int idx = blockIdx.x;
    int nt = idx & 3; idx >>= 2;
    int mt = idx % 192;
    int e  = idx / 192;
    const float* equ = e ? equ2 : equ1;

    int tid = threadIdx.x;
    int lane = tid & 31, wave = tid >> 5;
    int half = lane >> 4, ln = lane & 15;

    v8f acc[4] = {};

    for (int kc = 0; kc < 8; ++kc) {
        int k0 = kc * 32;
        {   // A tile: 128 x 32 f32 -> bf16 pairs in LDS
            int row = tid >> 1, part = tid & 1;
            const float* src = equ + (size_t)(mt * 128 + row) * 256 + k0 + part * 16;
            unsigned int* dst = &a_lds[row * 16 + part * 8];
#pragma unroll
            for (int i = 0; i < 4; ++i) {
                float4 f = ((const float4*)src)[i];
                dst[i * 2 + 0] = bf16x2(f.x, f.y);
                dst[i * 2 + 1] = bf16x2(f.z, f.w);
            }
        }
        {   // B tile: Wv rows o (64) x k (32); B[k][o] = Wv[o][k], stored col-major
#pragma unroll
            for (int i = 0; i < 2; ++i) {
                int fid = tid * 2 + i;
                int orow = fid >> 3, fpart = fid & 7;
                float4 f = *(const float4*)(Wv + (size_t)(nt * 64 + orow) * 256 + k0 + fpart * 4);
                b_lds[orow * 16 + fpart * 2 + 0] = bf16x2(f.x, f.y);
                b_lds[orow * 16 + fpart * 2 + 1] = bf16x2(f.z, f.w);
            }
        }
        __syncthreads();
        v16bf a = gather_frag(&a_lds[(wave * 16 + ln) * 16], half);
        v16bf bf[4];
#pragma unroll
        for (int ct = 0; ct < 4; ++ct)
            bf[ct] = gather_frag(&b_lds[(ct * 16 + ln) * 16], half);
#pragma unroll
        for (int ct = 0; ct < 4; ++ct)
            acc[ct] = __builtin_amdgcn_wmma_f32_16x16x32_bf16(
                false, a, false, bf[ct], (short)0, acc[ct], false, false);
        __syncthreads();
    }

    // Writeback: scatter into (e,b,h) 1024x96 bf16 layout.
    // idx = (((e*8+b)*8+h)*1024+m)*96 + l*32 + c
#pragma unroll
    for (int r = 0; r < 8; ++r) {
        int rr = mt * 128 + wave * 16 + half * 8 + r;  // flat (b,m,l)
        int q = rr / 3;                                 // = b*1024 + m
        int b_idx = q >> 10;
        int m = q & 1023;
        int l = rr - q * 3;
        size_t rowbase = (((size_t)e * 8 + b_idx) * 8) * (1024 * 96)
                       + (size_t)m * 96 + l * 32;
#pragma unroll
        for (int ct = 0; ct < 4; ++ct) {
            int o = nt * 64 + ct * 16 + ln;  // output channel 0..255
            int h = o >> 5, c = o & 31;
            v_ws[rowbase + (size_t)h * (1024 * 96) + c] =
                (unsigned short)bf16_1(acc[ct][r]);
        }
    }
}

// ---------------------------------------------------------------------------
// Kernel B: per (e,b,h): out(1024x96) = att(1024x1024, f32) x V(1024x96, bf16)
// Block handles 128 rows x all 96 cols -> att is read from HBM exactly once.
// Double-buffered LDS: global loads for chunk kc+1 issue before the chunk-kc
// WMMAs, convert+store lands after, one barrier per iteration.
// Result scattered to ao_ws[e][b][n][l][h*32+c] (bf16) for the Wo GEMM.
// ---------------------------------------------------------------------------
__global__ __launch_bounds__(256) void kB_attn(
    const float* __restrict__ att, const unsigned short* __restrict__ v_ws,
    unsigned short* __restrict__ ao_ws)
{
    __shared__ unsigned int   a_lds[2][128 * 16];   // att tile, bf16 pairs
    __shared__ unsigned short v_lds[2][96 * 32];    // V tile, [col][k]

    int idx = blockIdx.x;
    int nt = idx & 7;  idx >>= 3;
    int h  = idx & 7;  idx >>= 3;
    int b  = idx & 7;
    int e  = idx >> 3;

    const float* attp = att + (((size_t)e * 8 + b) * 8 + h) * 1024 * 1024;
    const unsigned short* vp = v_ws + (((size_t)e * 8 + b) * 8 + h) * 1024 * 96;

    int tid = threadIdx.x;
    int lane = tid & 31, wave = tid >> 5;
    int half = lane >> 4, ln = lane & 15;
    int n0 = nt * 128;

    // per-thread load geometry
    int arow  = tid >> 1, apart = tid & 1;            // att: 2 threads/row, 4 float4 each
    int krow  = tid >> 3, cp0   = (tid & 7) * 6;      // V: 8 threads/k-row, 6 dwords each
    const float* asrc0 = attp + (size_t)(n0 + arow) * 1024 + apart * 16;

    float4       aReg[4];
    unsigned int vReg[6];

    auto load_gl = [&](int kc) {
        const float* srcA = asrc0 + kc * 32;
#pragma unroll
        for (int i = 0; i < 4; ++i) aReg[i] = ((const float4*)srcA)[i];
        const unsigned int* srcV =
            (const unsigned int*)(vp + (size_t)(kc * 32 + krow) * 96);
#pragma unroll
        for (int i = 0; i < 6; ++i) vReg[i] = srcV[cp0 + i];
    };
    auto store_lds = [&](int buf) {
        unsigned int* dst = &a_lds[buf][arow * 16 + apart * 8];
#pragma unroll
        for (int i = 0; i < 4; ++i) {
            dst[i * 2 + 0] = bf16x2(aReg[i].x, aReg[i].y);
            dst[i * 2 + 1] = bf16x2(aReg[i].z, aReg[i].w);
        }
#pragma unroll
        for (int i = 0; i < 6; ++i) {
            unsigned int u = vReg[i];
            int c0 = (cp0 + i) * 2;
            v_lds[buf][c0 * 32 + krow]       = (unsigned short)(u & 0xFFFFu);
            v_lds[buf][(c0 + 1) * 32 + krow] = (unsigned short)(u >> 16);
        }
    };

    v8f acc[6] = {};

    load_gl(0);
    store_lds(0);
    __syncthreads();

    for (int kc = 0; kc < 32; ++kc) {
        int cur = kc & 1;
        if (kc + 1 < 32) load_gl(kc + 1);   // start next chunk's HBM traffic

        v16bf a = gather_frag(&a_lds[cur][(wave * 16 + ln) * 16], half);
        v16bf bf[6];
#pragma unroll
        for (int ct = 0; ct < 6; ++ct)
            bf[ct] = gather_frag(
                (const unsigned int*)&v_lds[cur][(ct * 16 + ln) * 32], half);
#pragma unroll
        for (int ct = 0; ct < 6; ++ct)
            acc[ct] = __builtin_amdgcn_wmma_f32_16x16x32_bf16(
                false, a, false, bf[ct], (short)0, acc[ct], false, false);

        if (kc + 1 < 32) store_lds((kc + 1) & 1);
        __syncthreads();
    }

    // Writeback to ao_ws[e][b][n][l][256] bf16: dst = ((eb*1024+n)*3+l)*256 + h*32 + c
#pragma unroll
    for (int r = 0; r < 8; ++r) {
        int n = n0 + wave * 16 + half * 8 + r;
        size_t rowbase = (((size_t)e * 8 + b) * 1024 + n) * 768 + h * 32;
#pragma unroll
        for (int ct = 0; ct < 6; ++ct) {
            int col = ct * 16 + ln;          // 0..95 = l*32 + c
            int l = col >> 5, c = col & 31;
            ao_ws[rowbase + l * 256 + c] = (unsigned short)bf16_1(acc[ct][r]);
        }
    }
}

// ---------------------------------------------------------------------------
// Kernel C: out = f32(attnout_bf16 @ Wo^T) + equ (residual). Same tiling as kA.
// ---------------------------------------------------------------------------
__global__ __launch_bounds__(256) void kC_out(
    const unsigned short* __restrict__ ao_ws, const float* __restrict__ Wo,
    const float* __restrict__ equ1, const float* __restrict__ equ2,
    float* __restrict__ out)
{
    __shared__ unsigned int a_lds[128 * 16];
    __shared__ unsigned int b_lds[64 * 16];

    int idx = blockIdx.x;
    int nt = idx & 3; idx >>= 2;
    int mt = idx % 192;
    int e  = idx / 192;
    const float* equ = e ? equ2 : equ1;
    const unsigned short* ap = ao_ws + (size_t)e * 24576 * 256;

    int tid = threadIdx.x;
    int lane = tid & 31, wave = tid >> 5;
    int half = lane >> 4, ln = lane & 15;

    v8f acc[4] = {};

    for (int kc = 0; kc < 8; ++kc) {
        int k0 = kc * 32;
        {   // A tile already bf16, k-pairs contiguous: straight copy
            int row = tid >> 1, part = tid & 1;
            const unsigned int* src =
                (const unsigned int*)(ap + (size_t)(mt * 128 + row) * 256 + k0) + part * 8;
            unsigned int* dst = &a_lds[row * 16 + part * 8];
#pragma unroll
            for (int i = 0; i < 8; ++i) dst[i] = src[i];
        }
        {   // Wo tile
#pragma unroll
            for (int i = 0; i < 2; ++i) {
                int fid = tid * 2 + i;
                int orow = fid >> 3, fpart = fid & 7;
                float4 f = *(const float4*)(Wo + (size_t)(nt * 64 + orow) * 256 + k0 + fpart * 4);
                b_lds[orow * 16 + fpart * 2 + 0] = bf16x2(f.x, f.y);
                b_lds[orow * 16 + fpart * 2 + 1] = bf16x2(f.z, f.w);
            }
        }
        __syncthreads();
        v16bf a = gather_frag(&a_lds[(wave * 16 + ln) * 16], half);
        v16bf bf[4];
#pragma unroll
        for (int ct = 0; ct < 4; ++ct)
            bf[ct] = gather_frag(&b_lds[(ct * 16 + ln) * 16], half);
#pragma unroll
        for (int ct = 0; ct < 4; ++ct)
            acc[ct] = __builtin_amdgcn_wmma_f32_16x16x32_bf16(
                false, a, false, bf[ct], (short)0, acc[ct], false, false);
        __syncthreads();
    }

    float* outp = out + (size_t)e * 24576 * 256;
#pragma unroll
    for (int r = 0; r < 8; ++r) {
        int rr = mt * 128 + wave * 16 + half * 8 + r;
        size_t rb = (size_t)rr * 256;
#pragma unroll
        for (int ct = 0; ct < 4; ++ct) {
            int o = nt * 64 + ct * 16 + ln;
            outp[rb + o] = acc[ct][r] + equ[rb + o];
        }
    }
}

extern "C" void kernel_launch(void* const* d_in, const int* in_sizes, int n_in,
                              void* d_out, int out_size, void* d_ws, size_t ws_size,
                              hipStream_t stream) {
    const float* att  = (const float*)d_in[0];  // (2,8,8,1024,1024) f32
    const float* equ1 = (const float*)d_in[1];  // (8,1024,3,256) f32
    const float* equ2 = (const float*)d_in[2];
    const float* Wv   = (const float*)d_in[3];  // (256,256) f32
    const float* Wo   = (const float*)d_in[4];
    float* out = (float*)d_out;                 // 2 x (8,1024,3,256) f32

    // Workspace: V (bf16, 24 MB) + attn-out (bf16, 24 MB) = 48 MB
    unsigned short* v_ws  = (unsigned short*)d_ws;
    unsigned short* ao_ws = v_ws + (size_t)2 * 8 * 8 * 1024 * 96;

    kA_v_gemm<<<1536, 256, 0, stream>>>(equ1, equ2, Wv, v_ws);
    kB_attn <<<1024, 256, 0, stream>>>(att, v_ws, ao_ws);
    kC_out  <<<1536, 256, 0, stream>>>(ao_ws, Wo, equ1, equ2, out);
}
